// GNN_57509612093941
// MI455X (gfx1250) — matrix operand order repaired and checked by reading, not compiled, not run on hardware
//
#include <hip/hip_runtime.h>

// ---------------------------------------------------------------------------
// GIN GNN forward for MI455X (gfx1250, wave32, WMMA bf16).
//   N=100000 nodes, E=1.6M edges, F=H=128, G=512 graphs, C=10, D=32
// d_out layout: [G*C log_softmax | G*D result3 | 1 mse_loss]
// ---------------------------------------------------------------------------

typedef __attribute__((ext_vector_type(16))) __bf16 v16bf;
typedef __attribute__((ext_vector_type(8)))  __bf16 v8bf;
typedef __attribute__((ext_vector_type(8)))  float  v8f;

#define HDIM 128
#define BN_EPS 1e-5f

__device__ __forceinline__ __bf16 tobf(float f) { return (__bf16)f; }

__device__ __forceinline__ v16bf cat8(v8bf lo, v8bf hi) {
  return __builtin_shufflevector(lo, hi, 0, 1, 2, 3, 4, 5, 6, 7,
                                         8, 9, 10, 11, 12, 13, 14, 15);
}

// ---- generic clear --------------------------------------------------------
__global__ void k_clear(float* __restrict__ p, int n) {
  int i = blockIdx.x * blockDim.x + threadIdx.x;
  if (i < n) p[i] = 0.0f;
}

// ---- agg = (1 + eps) * x  (float4 vectorized) -----------------------------
__global__ void k_agg_init(const float* __restrict__ x, const float* __restrict__ eps,
                           float* __restrict__ agg, int n4) {
  int i = blockIdx.x * blockDim.x + threadIdx.x;
  if (i < n4) {
    float s = 1.0f + eps[0];
    float4 v = ((const float4*)x)[i];
    v.x *= s; v.y *= s; v.z *= s; v.w *= s;
    ((float4*)agg)[i] = v;
  }
}

// ---- agg[dst] += x[src] : one wave32 per edge, 4 floats per lane ----------
__global__ void k_edge_scatter(const float* __restrict__ x, const int* __restrict__ src,
                               const int* __restrict__ dst, float* __restrict__ agg, int E) {
  long long t = (long long)blockIdx.x * blockDim.x + threadIdx.x;
  int e = (int)(t >> 5);
  int lane = (int)(t & 31);
  if (e < E) {
    int s = src[e], d = dst[e];
    float4 v = ((const float4*)(x + (size_t)s * HDIM))[lane];
    float* o = agg + (size_t)d * HDIM + lane * 4;
    atomicAdd(o + 0, v.x);
    atomicAdd(o + 1, v.y);
    atomicAdd(o + 2, v.z);
    atomicAdd(o + 3, v.w);
  }
}

// ---- Y[M,128] = relu(X[M,128] @ W[128,128] + b) via v_wmma_f32_16x16x32_bf16
// 256 threads = 8 waves; block computes a 128x128 tile; wave w owns rows
// w*16 .. w*16+15.  Activations staged row-major [m][k]; weights staged
// TRANSPOSED [n][k] so every WMMA fragment is two contiguous 16-byte runs in
// LDS -> ds_load_b128 pairs.  After compute, C tiles are re-laid-out through
// LDS (overlaying the dead sX/sWt space) so the global writeback is coalesced
// float4 stores instead of 64 divergent scalar stores per lane.
//
// A fragment (16-bit A, 16x32, ISA 7.12.2):
//   lane L: row M = L%16 ; element e -> K = (e/8)*16 + (L/16)*8 + (e%8)
// B fragment (32x16): lane L: col N = L%16 ; element e -> K = (e/8)*16 + (L/16)*8 + (e%8)
// C/D f32 16x16: element r -> row (L/16)*8 + r, col L%16
__global__ void __launch_bounds__(256)
k_gemm128(const float* __restrict__ X, const float* __restrict__ W,
          const float* __restrict__ b, float* __restrict__ Y, int M) {
  __shared__ __align__(16) char smem[65536];
  __bf16* sX  = (__bf16*)smem;              // [128][128] bf16 (32 KB)
  __bf16* sWt = (__bf16*)(smem + 32768);    // [128][128] bf16 transposed (32 KB)
  float*  sC  = (float*)smem;               // [128][128] f32 overlay after compute
  __shared__ float sB[128];

  int tid = threadIdx.x;
  int row0 = blockIdx.x * 128;

  if (tid < 128) sB[tid] = b[tid];
  // stage weights transposed: W row-major [K=128][N=128] -> sWt[n][k]
  for (int i = tid; i < 4096; i += 256) {
    float4 v = ((const float4*)W)[i];        // W[k][n..n+3]
    int k = i >> 5, n = (i & 31) << 2;
    sWt[(n + 0) * 128 + k] = tobf(v.x);
    sWt[(n + 1) * 128 + k] = tobf(v.y);
    sWt[(n + 2) * 128 + k] = tobf(v.z);
    sWt[(n + 3) * 128 + k] = tobf(v.w);
  }
  // stage activation block (zero-padded past M)
  for (int i = tid; i < 4096; i += 256) {
    int r = i >> 5;                           // 32 float4 per row
    int gr = row0 + r;
    float4 v = make_float4(0.f, 0.f, 0.f, 0.f);
    if (gr < M) v = ((const float4*)(X + (size_t)gr * HDIM))[i & 31];
    int j = i << 2;
    sX[j + 0] = tobf(v.x); sX[j + 1] = tobf(v.y);
    sX[j + 2] = tobf(v.z); sX[j + 3] = tobf(v.w);
  }
  // prefetch the next row-block of X toward L2 while this block computes
  if (row0 + 128 < M) {
    const float* nx = X + (size_t)(row0 + 128) * HDIM;
#pragma unroll
    for (int j = 0; j < 4; ++j)
      __builtin_prefetch(nx + (size_t)(tid + j * 256) * 16, 0, 1);
  }
  __syncthreads();

  int wave = tid >> 5, lane = tid & 31;
  int grp  = lane >> 4;                       // lane half 0/1
  int l16  = lane & 15;
  const __bf16* arow = sX + (wave * 16 + l16) * 128;
  int khalf = grp * 8;
  v16bf a[4];
#pragma unroll
  for (int kt = 0; kt < 4; ++kt)
    a[kt] = cat8(*(const v8bf*)(arow + kt * 32 + khalf),
                 *(const v8bf*)(arow + kt * 32 + 16 + khalf));

  int kb = grp * 8;
  v8f c[8];
#pragma unroll
  for (int nt = 0; nt < 8; ++nt) {
    const __bf16* brow = sWt + (nt * 16 + l16) * 128;
    c[nt] = (v8f){0.f, 0.f, 0.f, 0.f, 0.f, 0.f, 0.f, 0.f};
#pragma unroll
    for (int kt = 0; kt < 4; ++kt) {
      v16bf bf = cat8(*(const v8bf*)(brow + kt * 32 + kb),
                      *(const v8bf*)(brow + kt * 32 + 16 + kb));
      c[nt] = __builtin_amdgcn_wmma_f32_16x16x32_bf16(
                  /*neg_a=*/false, a[kt], /*neg_b=*/false, bf,
                  /*c_mod=*/(short)0, c[nt], /*reuse_a=*/false, /*reuse_b=*/false);
    }
  }
  __syncthreads();                             // all reads of sX/sWt complete

  // write C (+bias, ReLU) into LDS row-major f32 with immediate-offset stores
  int rbase = wave * 16 + grp * 8;
#pragma unroll
  for (int nt = 0; nt < 8; ++nt) {
    int ncol = nt * 16 + l16;
    float bias = sB[ncol];
    float* cp = sC + rbase * 128 + ncol;
#pragma unroll
    for (int r = 0; r < 8; ++r)
      cp[r * 128] = fmaxf(c[nt][r] + bias, 0.0f);
  }
  __syncthreads();

  // coalesced float4 writeback
  for (int i = tid; i < 4096; i += 256) {
    int r = i >> 5;
    int gr = row0 + r;
    if (gr < M)
      ((float4*)(Y + (size_t)gr * HDIM))[i & 31] = ((const float4*)sC)[i];
  }
}

// ---- BN stats: per-column sum & sumsq over all rows (atomics into stats) --
__global__ void k_bn_stats(const float* __restrict__ h, float* __restrict__ stats, int N) {
  __shared__ float ls[128], lq[128];
  int col  = threadIdx.x & 127;
  int half = threadIdx.x >> 7;            // 0 or 1
  int row0 = blockIdx.x * 128;
  int rend = row0 + 128; if (rend > N) rend = N;
  float s = 0.0f, q = 0.0f;
  for (int r = row0 + half; r < rend; r += 2) {
    float v = h[(size_t)r * HDIM + col];
    s += v; q += v * v;
  }
  if (half == 1) { ls[col] = s; lq[col] = q; }
  __syncthreads();
  if (half == 0) {
    s += ls[col]; q += lq[col];
    atomicAdd(&stats[col], s);
    atomicAdd(&stats[128 + col], q);
  }
}

// ---- BN apply (in place): h = gamma*(h-mu)*rsqrt(var+eps)+beta ------------
__global__ void k_bn_apply(float* __restrict__ h, const float* __restrict__ stats,
                           const float* __restrict__ gamma, const float* __restrict__ beta,
                           int N) {
  size_t i = (size_t)blockIdx.x * blockDim.x + threadIdx.x;
  size_t total = (size_t)N * HDIM;
  if (i < total) {
    int col = (int)(i & 127);
    float invN = 1.0f / (float)N;
    float mu  = stats[col] * invN;
    float var = stats[128 + col] * invN - mu * mu;
    float rs  = rsqrtf(var + BN_EPS);
    h[i] = gamma[col] * (h[i] - mu) * rs + beta[col];
  }
}

// ---- MSE: sum over nodes of (h@Wloss + bloss - degree)^2 -> acc -----------
__global__ void k_mse(const float* __restrict__ h, const float* __restrict__ Wl,
                      const float* __restrict__ bl, const float* __restrict__ deg,
                      float* __restrict__ acc, int N) {
  __shared__ float sw[128];
  __shared__ float red[256];
  int t = threadIdx.x;
  if (t < 128) sw[t] = Wl[t];
  __syncthreads();
  int n = blockIdx.x * 256 + t;
  float d = 0.0f;
  if (n < N) {
    float dot = bl[0];
    const float* hr = h + (size_t)n * HDIM;
#pragma unroll 8
    for (int k = 0; k < 128; ++k) dot += hr[k] * sw[k];
    float e = dot - deg[n];
    d = e * e;
  }
  red[t] = d;
  __syncthreads();
  for (int s = 128; s > 0; s >>= 1) {
    if (t < s) red[t] += red[t + s];
    __syncthreads();
  }
  if (t == 0) atomicAdd(acc, red[0]);
}

__global__ void k_mse_final(const float* __restrict__ acc, float* __restrict__ out, int N) {
  if (threadIdx.x == 0) out[0] = acc[0] / (float)N;
}

// ---- mean pool: pool[batch[n]] += h[n], cnt[batch[n]] += 1 ----------------
__global__ void k_pool(const float* __restrict__ h, const int* __restrict__ batch,
                       float* __restrict__ pool, float* __restrict__ cnts, int N) {
  long long t = (long long)blockIdx.x * blockDim.x + threadIdx.x;
  int n = (int)(t >> 5);
  int lane = (int)(t & 31);
  if (n < N) {
    int b = batch[n];
    float4 v = ((const float4*)(h + (size_t)n * HDIM))[lane];
    float* o = pool + (size_t)b * HDIM + lane * 4;
    atomicAdd(o + 0, v.x);
    atomicAdd(o + 1, v.y);
    atomicAdd(o + 2, v.z);
    atomicAdd(o + 3, v.w);
    if (lane == 0) atomicAdd(&cnts[b], 1.0f);
  }
}

__global__ void k_pool_div(float* __restrict__ pool, const float* __restrict__ cnts, int n) {
  int i = blockIdx.x * blockDim.x + threadIdx.x;
  if (i < n) pool[i] /= fmaxf(cnts[i >> 7], 1.0f);
}

// ---- head: per graph, logits->log_softmax (C=10) and result3 (D=32) -------
__global__ void k_head(const float* __restrict__ z, const float* __restrict__ W2,
                       const float* __restrict__ b2, const float* __restrict__ W4,
                       const float* __restrict__ b4, float* __restrict__ out1,
                       float* __restrict__ out3) {
  __shared__ float zr[128];
  __shared__ float lg[10];
  __shared__ float lse;
  int g = blockIdx.x, t = threadIdx.x;   // 64 threads = 2 waves
  zr[t]      = z[(size_t)g * HDIM + t];
  zr[t + 64] = z[(size_t)g * HDIM + t + 64];
  __syncthreads();
  if (t < 32) {                          // wave 0: result3
    float acc = b4[t];
#pragma unroll 8
    for (int k = 0; k < 128; ++k) acc += zr[k] * W4[k * 32 + t];
    out3[(size_t)g * 32 + t] = acc;
  } else if (t < 42) {                   // wave 1 lanes 0..9: logits
    int c = t - 32;
    float acc = b2[c];
#pragma unroll 8
    for (int k = 0; k < 128; ++k) acc += zr[k] * W2[k * 10 + c];
    lg[c] = acc;
  }
  __syncthreads();
  if (t == 0) {
    float mx = lg[0];
    for (int c = 1; c < 10; ++c) mx = fmaxf(mx, lg[c]);
    float s = 0.0f;
    for (int c = 0; c < 10; ++c) s += __expf(lg[c] - mx);
    lse = mx + __logf(s);
  }
  __syncthreads();
  if (t < 10) out1[(size_t)g * 10 + t] = lg[t] - lse;
}

// ---------------------------------------------------------------------------
extern "C" void kernel_launch(void* const* d_in, const int* in_sizes, int n_in,
                              void* d_out, int out_size, void* d_ws, size_t ws_size,
                              hipStream_t stream) {
  const float* x      = (const float*)d_in[0];
  const int*   ei     = (const int*)  d_in[1];
  const int*   batch  = (const int*)  d_in[2];
  const float* degree = (const float*)d_in[3];
  const float* eps1   = (const float*)d_in[4];
  const float* W1a    = (const float*)d_in[5];
  const float* b1a    = (const float*)d_in[6];
  const float* W1b    = (const float*)d_in[7];
  const float* b1b    = (const float*)d_in[8];
  const float* g1     = (const float*)d_in[9];
  const float* be1    = (const float*)d_in[10];
  const float* eps2   = (const float*)d_in[11];
  const float* W2a    = (const float*)d_in[12];
  const float* b2a    = (const float*)d_in[13];
  const float* W2b    = (const float*)d_in[14];
  const float* b2b    = (const float*)d_in[15];
  const float* g2     = (const float*)d_in[16];
  const float* be2    = (const float*)d_in[17];
  const float* Wlin1  = (const float*)d_in[18];
  const float* blin1  = (const float*)d_in[19];
  const float* Wlin2  = (const float*)d_in[20];
  const float* blin2  = (const float*)d_in[21];
  const float* Wlin4  = (const float*)d_in[22];
  const float* blin4  = (const float*)d_in[23];
  const float* Wloss  = (const float*)d_in[24];
  const float* bloss  = (const float*)d_in[25];

  const int N = in_sizes[2];
  const int E = in_sizes[1] / 2;
  const int G = (out_size - 1) / 42;     // C=10 + D=32 per graph
  const int* src = ei;
  const int* dst = ei + E;

  // workspace layout
  float* ws = (float*)d_ws;
  size_t NH = (size_t)N * HDIM;
  float* agg   = ws;
  float* tmpA  = agg + NH;
  float* h     = tmpA + NH;
  float* stats = h + NH;                       // 256 floats (sum|sumsq)
  float* pool  = stats + 256;                  // G*128
  float* cnts  = pool + (size_t)G * HDIM;      // G
  float* z     = cnts + G;                     // G*128
  float* acc   = z + (size_t)G * HDIM;         // 1

  float* out1 = (float*)d_out;                 // [G,10]
  float* out3 = out1 + (size_t)G * 10;         // [G,32]
  float* outm = out3 + (size_t)G * 32;         // scalar

  const int n4     = (int)(NH / 4);
  const dim3 blk(256);
  const int gAgg   = (n4 + 255) / 256;
  const int gEdge  = (int)(((long long)E * 32 + 255) / 256);
  const int gGemm  = (N + 127) / 128;
  const int gApply = (int)((NH + 255) / 256);
  const int gStats = (N + 127) / 128;
  const int gPoolG = (G + 127) / 128;

  // ===== GIN layer 1 =====
  k_agg_init    <<<gAgg,  blk, 0, stream>>>(x, eps1, agg, n4);
  k_edge_scatter<<<gEdge, blk, 0, stream>>>(x, src, dst, agg, E);
  k_gemm128     <<<gGemm, blk, 0, stream>>>(agg, W1a, b1a, tmpA, N);
  k_gemm128     <<<gGemm, blk, 0, stream>>>(tmpA, W1b, b1b, h, N);
  k_clear       <<<1, 256, 0, stream>>>(stats, 256);
  k_bn_stats    <<<gStats, blk, 0, stream>>>(h, stats, N);
  k_bn_apply    <<<gApply, blk, 0, stream>>>(h, stats, g1, be1, N);

  // ===== GIN layer 2 =====
  k_agg_init    <<<gAgg,  blk, 0, stream>>>(h, eps2, agg, n4);
  k_edge_scatter<<<gEdge, blk, 0, stream>>>(h, src, dst, agg, E);
  k_gemm128     <<<gGemm, blk, 0, stream>>>(agg, W2a, b2a, tmpA, N);
  k_gemm128     <<<gGemm, blk, 0, stream>>>(tmpA, W2b, b2b, h, N);
  k_clear       <<<1, 256, 0, stream>>>(stats, 256);
  k_bn_stats    <<<gStats, blk, 0, stream>>>(h, stats, N);
  k_bn_apply    <<<gApply, blk, 0, stream>>>(h, stats, g2, be2, N);

  // ===== auxiliary MSE loss =====
  k_clear       <<<1, 32, 0, stream>>>(acc, 1);
  k_mse         <<<(N + 255) / 256, blk, 0, stream>>>(h, Wloss, bloss, degree, acc, N);
  k_mse_final   <<<1, 32, 0, stream>>>(acc, outm, N);

  // ===== global mean pool =====
  k_clear       <<<(G * HDIM + G + 255) / 256, blk, 0, stream>>>(pool, G * HDIM + G);
  k_pool        <<<(int)(((long long)N * 32 + 255) / 256), blk, 0, stream>>>(h, batch, pool, cnts, N);
  k_pool_div    <<<(G * HDIM + 255) / 256, blk, 0, stream>>>(pool, cnts, G * HDIM);

  // ===== classifier head =====
  k_gemm128     <<<gPoolG, blk, 0, stream>>>(pool, Wlin1, blin1, z, G);
  k_head        <<<G, 64, 0, stream>>>(z, Wlin2, blin2, Wlin4, blin4, out1, out3);
}
